// attention_83932250898666
// MI455X (gfx1250) — compile-verified, matrix operand order checked by reading, not run on hardware
//
#include <hip/hip_runtime.h>
#include <hip/hip_bf16.h>

typedef __attribute__((ext_vector_type(16))) _Float16 v16h;
typedef __attribute__((ext_vector_type(8)))  _Float16 v8h;
typedef __attribute__((ext_vector_type(8)))  float    v8f;
typedef int v4i __attribute__((vector_size(16)));

#define B_ 4
#define N_ 2048
#define D_ 1024

#define BM 128
#define BN 128
#define BK 32

// CDNA5 async global->LDS path (ASYNCcnt-tracked, no VGPR round trip).
// Host pass / older toolchains fall back to the synchronous LDS copy.
#if defined(__HIP_DEVICE_COMPILE__) && __has_builtin(__builtin_amdgcn_global_load_async_to_lds_b128)
#define ASYNC_LDS 1
#else
#define ASYNC_LDS 0
#endif

#if ASYNC_LDS
typedef __attribute__((address_space(1))) v4i* gbl_v4i_p;   // __device__ (global)
typedef __attribute__((address_space(3))) v4i* lds_v4i_p;   // __shared__ (LDS)

__device__ inline gbl_v4i_p as_gbl(const void* p) {
    return (gbl_v4i_p)(uintptr_t)p;
}
__device__ inline lds_v4i_p as_lds(const void* p) {
    // generic LDS address: addr[31:0] is the LDS offset (ISA 10.2 aperture rule)
    return (lds_v4i_p)(uint32_t)(uintptr_t)p;
}
__device__ inline void wait_async0() {
#if __has_builtin(__builtin_amdgcn_s_wait_asynccnt)
    __builtin_amdgcn_s_wait_asynccnt(0);
#else
    asm volatile("s_wait_asynccnt 0x0" ::: "memory");
#endif
}
#endif

// ---------------------------------------------------------------------------
// fp32 -> fp16 flat cast (4 elems/thread)
// ---------------------------------------------------------------------------
__global__ __launch_bounds__(256)
void cast_f16_kernel(const float* __restrict__ src, _Float16* __restrict__ dst, size_t n) {
    size_t i = ((size_t)blockIdx.x * blockDim.x + threadIdx.x) * 4;
    if (i + 3 < n) {
        dst[i + 0] = (_Float16)src[i + 0];
        dst[i + 1] = (_Float16)src[i + 1];
        dst[i + 2] = (_Float16)src[i + 2];
        dst[i + 3] = (_Float16)src[i + 3];
    }
}

// ---------------------------------------------------------------------------
// Transpose the three projection weights to f16: T[k][d] = W[d][k]
// ---------------------------------------------------------------------------
__global__ __launch_bounds__(256)
void transpose_w3_kernel(const float* __restrict__ Wk, const float* __restrict__ Wq,
                         const float* __restrict__ Wv,
                         _Float16* __restrict__ Tk, _Float16* __restrict__ Tq,
                         _Float16* __restrict__ Tv, int Dd) {
    int k = blockIdx.x * 16 + (threadIdx.x & 15);
    int d = blockIdx.y * 16 + (threadIdx.x >> 4);
    const float* W = (blockIdx.z == 0) ? Wk : (blockIdx.z == 1) ? Wq : Wv;
    _Float16*    T = (blockIdx.z == 0) ? Tk : (blockIdx.z == 1) ? Tq : Tv;
    if (k < Dd && d < Dd)
        T[(size_t)k * Dd + d] = (_Float16)W[(size_t)d * Dd + k];
}

// ---------------------------------------------------------------------------
// Load one 16x32 f16 operand fragment from an LDS tile (row-major, ld = BK).
// Lane L (g = L/16, r = L%16): halves 0..7 -> K=8g.., halves 8..15 -> K=16+8g..
// ---------------------------------------------------------------------------
__device__ inline v16h ld_frag(const _Float16* __restrict__ s, int row, int g, int r) {
    const _Float16* p = s + (size_t)(row + r) * BK;
    v8h lo = *(const v8h*)(p + 8 * g);
    v8h hi = *(const v8h*)(p + 16 + 8 * g);
    return __builtin_shufflevector(lo, hi, 0, 1, 2, 3, 4, 5, 6, 7,
                                           8, 9, 10, 11, 12, 13, 14, 15);
}

// ---------------------------------------------------------------------------
// Tiled WMMA GEMM:  C[m][n] = sum_k A[m][k] * Bm[n][k]   (A * Bm^T)
// MODE 0: f16 out row-major + bias ; MODE 1: f16 out transposed + bias ;
// MODE 2: f32 out row-major.
// 256 threads = 8 waves (2 x 4); wave tile 64x32 = 4x2 WMMA accumulators.
// Double-buffered LDS tiles staged with GLOBAL_LOAD_ASYNC_TO_LDS_B128.
// ---------------------------------------------------------------------------
template <int MODE>
__global__ __launch_bounds__(256)
void gemm_wmma_kernel(const _Float16* __restrict__ A, int ldA, size_t bsA,
                      const _Float16* __restrict__ Bm, int ldB, size_t bsB,
                      const float* __restrict__ bias,
                      _Float16* __restrict__ outH, float* __restrict__ outF,
                      int ldC, size_t bsC, int K) {
    __shared__ _Float16 As[2][BM * BK];  // 2 x 8 KB
    __shared__ _Float16 Bs[2][BN * BK];  // 2 x 8 KB

    const int tid  = threadIdx.x;
    const int lane = tid & 31;
    const int wv   = tid >> 5;
    const int wm0  = (wv & 1) * 64;
    const int wn0  = (wv >> 1) * 32;
    const int g    = lane >> 4;
    const int r    = lane & 15;

    const size_t bm = (size_t)blockIdx.x * BM;
    const size_t bn = (size_t)blockIdx.y * BN;

    const _Float16* Ab = A  + (size_t)blockIdx.z * bsA + bm * (size_t)ldA;
    const _Float16* Bb = Bm + (size_t)blockIdx.z * bsB + bn * (size_t)ldB;

    // Each thread stages two 16B chunks per tile (branchless): chunk ids
    // tid and tid+256 over a 128-row x 4-chunk tile.
    const int r0 = tid >> 2, o0 = (tid & 3) * 8;
    const int r1 = r0 + 64;             // (tid + 256) >> 2
    const _Float16* ga0 = Ab + (size_t)r0 * ldA + o0;
    const _Float16* ga1 = Ab + (size_t)r1 * ldA + o0;
    const _Float16* gb0 = Bb + (size_t)r0 * ldB + o0;
    const _Float16* gb1 = Bb + (size_t)r1 * ldB + o0;
    const int so0 = r0 * BK + o0;
    const int so1 = r1 * BK + o0;

    auto stage = [&](int buf, int kt) {
#if ASYNC_LDS
        __builtin_amdgcn_global_load_async_to_lds_b128(as_gbl(ga0 + kt), as_lds(&As[buf][so0]), 0, 0);
        __builtin_amdgcn_global_load_async_to_lds_b128(as_gbl(ga1 + kt), as_lds(&As[buf][so1]), 0, 0);
        __builtin_amdgcn_global_load_async_to_lds_b128(as_gbl(gb0 + kt), as_lds(&Bs[buf][so0]), 0, 0);
        __builtin_amdgcn_global_load_async_to_lds_b128(as_gbl(gb1 + kt), as_lds(&Bs[buf][so1]), 0, 0);
#else
        *(v8h*)&As[buf][so0] = *(const v8h*)(ga0 + kt);
        *(v8h*)&As[buf][so1] = *(const v8h*)(ga1 + kt);
        *(v8h*)&Bs[buf][so0] = *(const v8h*)(gb0 + kt);
        *(v8h*)&Bs[buf][so1] = *(const v8h*)(gb1 + kt);
#endif
    };

    v8f acc[4][2];
#pragma unroll
    for (int i = 0; i < 4; ++i)
#pragma unroll
        for (int j = 0; j < 2; ++j)
#pragma unroll
            for (int v = 0; v < 8; ++v) acc[i][j][v] = 0.0f;

    const int nk = K / BK;
    stage(0, 0);

    for (int t = 0; t < nk; ++t) {
        const int cur = t & 1;
#if ASYNC_LDS
        wait_async0();                  // own async contributions done
#endif
        __syncthreads();                // => whole tile ready; prev reads drained
        if (t + 1 < nk) stage(cur ^ 1, (t + 1) * BK);

        v16h af[4], bf[2];
#pragma unroll
        for (int j = 0; j < 2; ++j) bf[j] = ld_frag(Bs[cur], wn0 + j * 16, g, r);
#pragma unroll
        for (int i = 0; i < 4; ++i) af[i] = ld_frag(As[cur], wm0 + i * 16, g, r);

#pragma unroll
        for (int i = 0; i < 4; ++i)
#pragma unroll
            for (int j = 0; j < 2; ++j)
                acc[i][j] = __builtin_amdgcn_wmma_f32_16x16x32_f16(
                    false, af[i], false, bf[j], (short)0, acc[i][j], false, false);
    }

    // Epilogue: fragment element (vgpr v, lane) -> row m = v + 8g, col n = r.
    float*    outFb = outF + (MODE == 2 ? (size_t)blockIdx.z * bsC : 0);
    _Float16* outHb = outH + (MODE != 2 ? (size_t)blockIdx.z * bsC : 0);
#pragma unroll
    for (int i = 0; i < 4; ++i) {
#pragma unroll
        for (int j = 0; j < 2; ++j) {
            size_t col = bn + wn0 + j * 16 + r;
            float  bv  = (MODE == 2) ? 0.0f : bias[col];
#pragma unroll
            for (int v = 0; v < 8; ++v) {
                size_t row = bm + wm0 + i * 16 + v + 8 * g;
                float  val = acc[i][j][v];
                if (MODE == 0)
                    outHb[row * (size_t)ldC + col] = (_Float16)(val + bv);
                else if (MODE == 1)
                    outHb[col * (size_t)ldC + row] = (_Float16)(val + bv);
                else
                    outFb[row * (size_t)ldC + col] = val;
            }
        }
    }
}

// ---------------------------------------------------------------------------
// Row softmax (in place on fp32 scores living in d_out's `weight` slot),
// emitting an f16 copy for the att GEMM. scale = 1/sqrt(D).
// ---------------------------------------------------------------------------
__global__ __launch_bounds__(256)
void softmax_kernel(float* __restrict__ w, _Float16* __restrict__ w16,
                    int N, float scale) {
    __shared__ float red[256];
    const size_t base = (size_t)blockIdx.x * N;
    const int    tid  = threadIdx.x;

    float m = -3.402823466e38f;
    for (int i = tid; i < N; i += 256) m = fmaxf(m, w[base + i] * scale);
    red[tid] = m;
    __syncthreads();
    for (int off = 128; off > 0; off >>= 1) {
        if (tid < off) red[tid] = fmaxf(red[tid], red[tid + off]);
        __syncthreads();
    }
    m = red[0];
    __syncthreads();

    float sum = 0.0f;
    for (int i = tid; i < N; i += 256) {
        float e = __expf(w[base + i] * scale - m);
        w[base + i] = e;
        sum += e;
    }
    red[tid] = sum;
    __syncthreads();
    for (int off = 128; off > 0; off >>= 1) {
        if (tid < off) red[tid] += red[tid + off];
        __syncthreads();
    }
    float inv = 1.0f / red[0];

    for (int i = tid; i < N; i += 256) {
        float p = w[base + i] * inv;
        __builtin_nontemporal_store(p, &w[base + i]);  // final output, not re-read
        w16[base + i] = (_Float16)p;                   // re-read by att GEMM: keep RT
    }
}

// ---------------------------------------------------------------------------
// out1 = LayerNorm(x + att) * gamma + beta ; out = concat(x, out1).
// ---------------------------------------------------------------------------
__global__ __launch_bounds__(256)
void ln_concat_kernel(const float* __restrict__ x, const float* __restrict__ att,
                      const float* __restrict__ gamma, const float* __restrict__ beta,
                      float* __restrict__ out, int D) {
    __shared__ float red[256];
    __shared__ float hbuf[D_];
    const size_t row  = blockIdx.x;
    const float* xr   = x + row * (size_t)D;
    const float* ar   = att + row * (size_t)D;
    float*       outr = out + row * (size_t)(2 * D);
    const int    tid  = threadIdx.x;

    float s = 0.0f;
    for (int i = tid; i < D; i += 256) {
        float h = xr[i] + ar[i];
        hbuf[i] = h;
        s += h;
    }
    red[tid] = s;
    __syncthreads();
    for (int off = 128; off > 0; off >>= 1) {
        if (tid < off) red[tid] += red[tid + off];
        __syncthreads();
    }
    float mean = red[0] / (float)D;
    __syncthreads();

    float v = 0.0f;
    for (int i = tid; i < D; i += 256) {
        float d = hbuf[i] - mean;
        v += d * d;
    }
    red[tid] = v;
    __syncthreads();
    for (int off = 128; off > 0; off >>= 1) {
        if (tid < off) red[tid] += red[tid + off];
        __syncthreads();
    }
    float inv = rsqrtf(red[0] / (float)D + 1e-5f);

    for (int i = tid; i < D; i += 256) {
        __builtin_nontemporal_store(xr[i], &outr[i]);                       // write-only
        float o = (hbuf[i] - mean) * inv * gamma[i] + beta[i];
        __builtin_nontemporal_store(o, &outr[D + i]);                       // write-only
    }
}

// ---------------------------------------------------------------------------
extern "C" void kernel_launch(void* const* d_in, const int* in_sizes, int n_in,
                              void* d_out, int out_size, void* d_ws, size_t ws_size,
                              hipStream_t stream) {
    const float* x     = (const float*)d_in[0];
    const float* Wk    = (const float*)d_in[1];
    const float* bk    = (const float*)d_in[2];
    const float* Wq    = (const float*)d_in[3];
    const float* bq    = (const float*)d_in[4];
    const float* Wv    = (const float*)d_in[5];
    const float* bv    = (const float*)d_in[6];
    const float* gamma = (const float*)d_in[7];
    const float* beta  = (const float*)d_in[8];

    float* out = (float*)d_out;                       // [B,N,2D]
    float* att = out + (size_t)B_ * N_ * 2 * D_;      // [B,N,D]
    float* wgt = att + (size_t)B_ * N_ * D_;          // [B,N,N]

    const size_t BND = (size_t)B_ * N_ * D_;
    const size_t BNN = (size_t)B_ * N_ * N_;

    char* p = (char*)d_ws;
    _Float16* x16  = (_Float16*)p;  p += BND * 2;
    _Float16* wkT  = (_Float16*)p;  p += (size_t)D_ * D_ * 2;
    _Float16* wqT  = (_Float16*)p;  p += (size_t)D_ * D_ * 2;
    _Float16* wvT  = (_Float16*)p;  p += (size_t)D_ * D_ * 2;
    _Float16* q16  = (_Float16*)p;  p += BND * 2;
    _Float16* k16  = (_Float16*)p;  p += BND * 2;
    _Float16* vt16 = (_Float16*)p;  p += BND * 2;     // [D][B*N] (transposed)
    _Float16* w16  = (_Float16*)p;  p += BNN * 2;

    cast_f16_kernel<<<(unsigned)(BND / 1024), 256, 0, stream>>>(x, x16, BND);

    transpose_w3_kernel<<<dim3(D_ / 16, D_ / 16, 3), 256, 0, stream>>>(
        Wk, Wq, Wv, wkT, wqT, wvT, D_);

    dim3 gqkv((B_ * N_) / BM, D_ / BN, 1);
    gemm_wmma_kernel<0><<<gqkv, 256, 0, stream>>>(
        x16, D_, 0, wkT, D_, 0, bk, k16, nullptr, D_, 0, D_);
    gemm_wmma_kernel<0><<<gqkv, 256, 0, stream>>>(
        x16, D_, 0, wqT, D_, 0, bq, q16, nullptr, D_, 0, D_);
    gemm_wmma_kernel<1><<<gqkv, 256, 0, stream>>>(
        x16, D_, 0, wvT, D_, 0, bv, vt16, nullptr, B_ * N_, 0, D_);

    dim3 gsc(N_ / BM, N_ / BN, B_);
    gemm_wmma_kernel<2><<<gsc, 256, 0, stream>>>(
        q16, D_, (size_t)N_ * D_, k16, D_, (size_t)N_ * D_,
        nullptr, nullptr, wgt, N_, (size_t)N_ * N_, D_);

    softmax_kernel<<<B_ * N_, 256, 0, stream>>>(wgt, w16, N_, 0.03125f);

    dim3 gat(N_ / BM, D_ / BN, B_);
    gemm_wmma_kernel<2><<<gat, 256, 0, stream>>>(
        w16, N_, (size_t)N_ * N_, vt16, B_ * N_, (size_t)N_,
        nullptr, nullptr, att, D_, (size_t)N_ * D_, N_);

    ln_concat_kernel<<<B_ * N_, 256, 0, stream>>>(x, att, gamma, beta, out, D_);
}
